// PointTransformer_FP_78314433675655
// MI455X (gfx1250) — matrix operand order repaired and checked by reading, not compiled, not run on hardware
//
#include <hip/hip_runtime.h>
#include <hip/hip_bf16.h>

// ---------------------------------------------------------------------------
// PointTransformer forward for MI455X (gfx1250, wave32, WMMA).
// Generic stride-parameterized WMMA GEMM (f16 in / f32 accum,
// v_wmma_f32_16x16x32_f16) with fused BN/bias/ReLU/residual epilogue.
// Fast path: block = 4 waves computing 64x64 of Y; the 32x64 B tile is staged
// once per block in LDS as f16 (cooperative fill, ds_store_b128), fragments
// read back as ds_load_b128 pairs -> 4x less global B traffic and 4x fewer
// f32->f16 converts than per-wave loading. Edge path (stem K=3, head M=8):
// branch-free clamped loads.
// ---------------------------------------------------------------------------

typedef __attribute__((ext_vector_type(16))) _Float16 v16h;
typedef __attribute__((ext_vector_type(8)))  float    v8f;
typedef long long ll;

#define BNS_C 0.99999500003749977f   // 1/sqrt(1+1e-5)

__device__ __forceinline__ float epi(float val, int mm,
    const float* g, const float* bnb, const float* cb, int relu)
{
  float c0 = cb ? cb[mm] : 0.f;
  float s  = g  ? g[mm] * BNS_C : 1.f;
  float t  = bnb ? bnb[mm] : 0.f;
  val = (val + c0) * s + t;
  return relu ? fmaxf(val, 0.f) : val;
}

// ----------------- fast GEMM: 64x64 per block, LDS-staged B ----------------
__global__ void __launch_bounds__(128)
k_gemm_wmma_fast(
    const float* __restrict__ A, ll aB, ll aM, ll aK,
    const float* __restrict__ Bm, ll bB, ll bK, ll bN,
    float* __restrict__ Y, ll yB, ll yM, ll yN,
    int M, int N, int K,
    const float* __restrict__ g, const float* __restrict__ bnb,
    const float* __restrict__ cb,
    const float* __restrict__ res, ll resB, int relu)
{
  // LDS B tile: f16, [n_local 0..63][k 0..31], 4KB; uint4 = 8 halves.
  __shared__ uint4 lbs4[256];

  const int tid  = threadIdx.x;
  const int lane = tid & 31;
  const int wv   = tid >> 5;
  const int mt   = blockIdx.y * 4 + wv;
  const bool active = (mt * 16 < M);    // uniform per wave
  const int n0 = blockIdx.x * 64;
  const int b  = blockIdx.z;
  const int m0 = mt * 16;
  const int row = lane & 15;
  const int hi  = lane >> 4;

  const float* Ab    = A  + (ll)b * aB + (ll)(m0 + row) * aM;
  const float* Bbase = Bm + (ll)b * bB;

  // cooperative fill mapping: thread -> (n_local, k-halves fk..fk+15)
  const int fn = tid & 63;
  const int fk = (tid >> 6) * 16;
  const float* Bf = Bbase + (ll)(n0 + fn) * bN;

  v8f acc0 = {}, acc1 = {}, acc2 = {}, acc3 = {};

  for (int k0 = 0; k0 < K; k0 += 32) {
    __syncthreads();                       // previous tile fully consumed
    {
      union { _Float16 h[16]; uint4 u[2]; } w;
      if (bK == 1) {                       // contiguous -> b128 loads
        const float* p = Bf + k0 + fk;
        float4 u0 = ((const float4*)p)[0];
        float4 u1 = ((const float4*)p)[1];
        float4 u2 = ((const float4*)p)[2];
        float4 u3 = ((const float4*)p)[3];
        w.h[0]=(_Float16)u0.x;  w.h[1]=(_Float16)u0.y;  w.h[2]=(_Float16)u0.z;  w.h[3]=(_Float16)u0.w;
        w.h[4]=(_Float16)u1.x;  w.h[5]=(_Float16)u1.y;  w.h[6]=(_Float16)u1.z;  w.h[7]=(_Float16)u1.w;
        w.h[8]=(_Float16)u2.x;  w.h[9]=(_Float16)u2.y;  w.h[10]=(_Float16)u2.z; w.h[11]=(_Float16)u2.w;
        w.h[12]=(_Float16)u3.x; w.h[13]=(_Float16)u3.y; w.h[14]=(_Float16)u3.z; w.h[15]=(_Float16)u3.w;
      } else {
#pragma unroll
        for (int j = 0; j < 16; ++j)
          w.h[j] = (_Float16)Bf[(ll)(k0 + fk + j) * bK];
      }
      int ui = fn * 4 + (fk >> 3);
      lbs4[ui]     = w.u[0];
      lbs4[ui + 1] = w.u[1];
    }
    __syncthreads();                       // tile visible to all waves

    if (active) {
      const int kaBase = k0 + hi * 8;
      v16h af;
      if (aK == 1) {
        const float* Ap = Ab + kaBase;
        float4 u0 = *(const float4*)(Ap);
        float4 u1 = *(const float4*)(Ap + 4);
        float4 u2 = *(const float4*)(Ap + 16);
        float4 u3 = *(const float4*)(Ap + 20);
        af[0]=(_Float16)u0.x;  af[1]=(_Float16)u0.y;  af[2]=(_Float16)u0.z;  af[3]=(_Float16)u0.w;
        af[4]=(_Float16)u1.x;  af[5]=(_Float16)u1.y;  af[6]=(_Float16)u1.z;  af[7]=(_Float16)u1.w;
        af[8]=(_Float16)u2.x;  af[9]=(_Float16)u2.y;  af[10]=(_Float16)u2.z; af[11]=(_Float16)u2.w;
        af[12]=(_Float16)u3.x; af[13]=(_Float16)u3.y; af[14]=(_Float16)u3.z; af[15]=(_Float16)u3.w;
      } else {
#pragma unroll
        for (int hh = 0; hh < 8; ++hh) {
          af[hh]     = (_Float16)Ab[(ll)(kaBase + hh) * aK];
          af[hh + 8] = (_Float16)Ab[(ll)(kaBase + 16 + hh) * aK];
        }
      }
      union F { uint4 u[2]; v16h v; };
      F f0, f1, f2, f3;
      const int base = row * 4 + hi;
      f0.u[0] = lbs4[base];        f0.u[1] = lbs4[base + 2];
      f1.u[0] = lbs4[base + 64];   f1.u[1] = lbs4[base + 66];
      f2.u[0] = lbs4[base + 128];  f2.u[1] = lbs4[base + 130];
      f3.u[0] = lbs4[base + 192];  f3.u[1] = lbs4[base + 194];
      acc0 = __builtin_amdgcn_wmma_f32_16x16x32_f16(false, af, false, f0.v, (short)0, acc0, false, false);
      acc1 = __builtin_amdgcn_wmma_f32_16x16x32_f16(false, af, false, f1.v, (short)0, acc1, false, false);
      acc2 = __builtin_amdgcn_wmma_f32_16x16x32_f16(false, af, false, f2.v, (short)0, acc2, false, false);
      acc3 = __builtin_amdgcn_wmma_f32_16x16x32_f16(false, af, false, f3.v, (short)0, acc3, false, false);
    }
  }

  if (!active) return;
  float* Yb = Y + (ll)b * yB;
  const float* Rb = res ? (res + (ll)b * resB) : nullptr;
  v8f accs[4] = {acc0, acc1, acc2, acc3};
#pragma unroll
  for (int t = 0; t < 4; ++t) {
    int nn = n0 + t * 16 + row;
#pragma unroll
    for (int v = 0; v < 8; ++v) {
      int mm = m0 + v + 8 * hi;
      float val = epi(accs[t][v], mm, g, bnb, cb, relu);
      if (Rb) val += Rb[(ll)mm * yM + (ll)nn * yN];
      Yb[(ll)mm * yM + (ll)nn * yN] = val;
    }
  }
}

// ------------------- edge GEMM: clamped (branch-free) loads ----------------
__global__ void k_gemm_wmma_edge(
    const float* __restrict__ A, ll aB, ll aM, ll aK,
    const float* __restrict__ Bm, ll bB, ll bK, ll bN,
    float* __restrict__ Y, ll yB, ll yM, ll yN,
    int M, int N, int K,
    const float* __restrict__ g, const float* __restrict__ bnb,
    const float* __restrict__ cb,
    const float* __restrict__ res, ll resB, int relu)
{
  const int lane = threadIdx.x & 31;
  const int wv   = threadIdx.x >> 5;
  const int mt   = blockIdx.y * 4 + wv;
  if (mt * 16 >= M) return;
  const int nt = blockIdx.x;
  const int b  = blockIdx.z;
  const int m0 = mt * 16, n0 = nt * 16;
  const int row = lane & 15;
  const int hi  = lane >> 4;

  const float* Ab = A  + (ll)b * aB;
  const float* Bb = Bm + (ll)b * bB;
  const int mm = m0 + row;
  const int mmc = mm < M ? mm : (M - 1);

  v8f acc = {};
  for (int k0 = 0; k0 < K; k0 += 32) {
    v16h af, bf;
#pragma unroll
    for (int h = 0; h < 16; ++h) {
      int ka  = k0 + hi * 8 + (h & 7) + ((h >> 3) << 4);
      int kac = ka < K ? ka : (K - 1);
      float va = Ab[(ll)mmc * aM + (ll)kac * aK];
      if (ka >= K || mm >= M) va = 0.f;
      af[h] = (_Float16)va;
      int kb  = k0 + hi * 8 + (h & 7) + ((h >> 3) << 4);
      int kbc = kb < K ? kb : (K - 1);
      float vb = Bb[(ll)kbc * bK + (ll)(n0 + row) * bN];
      if (kb >= K) vb = 0.f;
      bf[h] = (_Float16)vb;
    }
    acc = __builtin_amdgcn_wmma_f32_16x16x32_f16(
        false, af, false, bf, (short)0, acc, false, false);
  }

  float* Yb = Y + (ll)b * yB;
  const float* Rb = res ? (res + (ll)b * resB) : nullptr;
#pragma unroll
  for (int v = 0; v < 8; ++v) {
    int m2 = m0 + v + 8 * hi;
    int nn = n0 + row;
    if (m2 >= M) continue;
    float val = epi(acc[v], m2, g, bnb, cb, relu);
    if (Rb) val += Rb[(ll)m2 * yM + (ll)nn * yN];
    Yb[(ll)m2 * yM + (ll)nn * yN] = val;
  }
}

static inline void gemm(hipStream_t st,
    const float* A, ll aB, ll aM, ll aK,
    const float* Bp, ll bB, ll bK, ll bN,
    float* Y, ll yB, ll yM, ll yN,
    int M, int N, int K, int batch,
    const float* g, const float* bnb, const float* cb,
    const float* res, ll resB, int relu)
{
  unsigned mtiles = (unsigned)((M + 15) / 16);
  bool fast = (K % 32 == 0) && (M % 16 == 0) && (N % 64 == 0);
  if (fast) {
    dim3 grd((unsigned)(N / 64), (mtiles + 3) / 4, (unsigned)batch);
    k_gemm_wmma_fast<<<grd, 128, 0, st>>>(A, aB, aM, aK, Bp, bB, bK, bN,
        Y, yB, yM, yN, M, N, K, g, bnb, cb, res, resB, relu);
  } else {
    dim3 grd((unsigned)(N / 16), (mtiles + 3) / 4, (unsigned)batch);
    k_gemm_wmma_edge<<<grd, 128, 0, st>>>(A, aB, aM, aK, Bp, bB, bK, bN,
        Y, yB, yM, yN, M, N, K, g, bnb, cb, res, resB, relu);
  }
}

// ------------------------- farthest point sampling -------------------------
__global__ void k_fps(const float* __restrict__ xyz, int Nn, int S,
                      int* __restrict__ out)
{
  __shared__ float dist[8192];
  __shared__ float rv[256];
  __shared__ int   ri[256];
  __shared__ int   far_s;
  const int b = blockIdx.x;
  const int t = threadIdx.x, T = blockDim.x;
  const float* P = xyz + (ll)b * Nn * 3;
  for (int i = t; i < Nn; i += T) dist[i] = 1e10f;
  if (t == 0) far_s = 0;
  __syncthreads();
  for (int it = 0; it < S; ++it) {
    int far = far_s;
    if (t == 0) out[(ll)b * S + it] = far;
    float cx = P[far*3], cy = P[far*3+1], cz = P[far*3+2];
    float bm = -1.f; int bi = 0;
    for (int i = t; i < Nn; i += T) {
      float dx = P[i*3]-cx, dy = P[i*3+1]-cy, dz = P[i*3+2]-cz;
      float d = fminf(dist[i], dx*dx + dy*dy + dz*dz);
      dist[i] = d;
      if (d > bm) { bm = d; bi = i; }
    }
    rv[t] = bm; ri[t] = bi;
    __syncthreads();
    for (int s2 = T >> 1; s2 > 0; s2 >>= 1) {
      if (t < s2 && rv[t + s2] > rv[t]) { rv[t] = rv[t + s2]; ri[t] = ri[t + s2]; }
      __syncthreads();
    }
    if (t == 0) far_s = ri[0];
    __syncthreads();
  }
}

// ------------------------------- kNN (top-k) -------------------------------
__global__ void k_knn(const float* __restrict__ xyz, const float* __restrict__ qxyz,
                      int Nn, int S, int Kk, int* __restrict__ idx)
{
  __shared__ float d[8192];
  __shared__ float rv[256];
  __shared__ int   ri[256];
  const int b = blockIdx.y, sq = blockIdx.x;
  const int t = threadIdx.x, T = blockDim.x;
  const float* P = xyz + (ll)b * Nn * 3;
  const float* Q = qxyz + ((ll)b * S + sq) * 3;
  float qx = Q[0], qy = Q[1], qz = Q[2];
  for (int i = t; i < Nn; i += T) {
    float dx = P[i*3]-qx, dy = P[i*3+1]-qy, dz = P[i*3+2]-qz;
    d[i] = dx*dx + dy*dy + dz*dz;
  }
  __syncthreads();
  for (int j = 0; j < Kk; ++j) {
    float bm = 1e30f; int bi = 0;
    for (int i = t; i < Nn; i += T) {
      float v = d[i];
      if (v < bm) { bm = v; bi = i; }
    }
    rv[t] = bm; ri[t] = bi;
    __syncthreads();
    for (int s2 = T >> 1; s2 > 0; s2 >>= 1) {
      if (t < s2 && rv[t + s2] < rv[t]) { rv[t] = rv[t + s2]; ri[t] = ri[t + s2]; }
      __syncthreads();
    }
    if (t == 0) { idx[((ll)b * S + sq) * Kk + j] = ri[0]; d[ri[0]] = 1e30f; }
    __syncthreads();
  }
}

// --------------------------- small helper kernels --------------------------
__global__ void k_gather_xyz(const float* __restrict__ xyz, const int* __restrict__ fidx,
                             int Nn, int S, int Bb, float* __restrict__ out)
{
  int i = blockIdx.x * blockDim.x + threadIdx.x;
  if (i >= Bb * S) return;
  int b = i / S, s = i % S;
  int id = fidx[(ll)b * S + s];
  const float* p = xyz + ((ll)b * Nn + id) * 3;
  float* o = out + ((ll)b * S + s) * 3;
  o[0] = p[0]; o[1] = p[1]; o[2] = p[2];
}

__global__ void k_group(const float* __restrict__ feat, const int* __restrict__ fidx,
                        const int* __restrict__ knn, int Nn, int S, int Kk, int D,
                        ll total, float* __restrict__ out)
{
  ll i = (ll)blockIdx.x * blockDim.x + threadIdx.x;
  if (i >= total) return;
  int c = (int)(i % D);
  ll r = i / D;
  int j = (int)(r % Kk); r /= Kk;
  int s = (int)(r % S);
  int b = (int)(r / S);
  int ctr = fidx[(ll)b * S + s];
  int nb  = knn[((ll)b * S + s) * Kk + j];
  float fc = feat[((ll)b * D + c) * Nn + ctr];
  float fn = feat[((ll)b * D + c) * Nn + nb];
  ll ro = (((ll)b * S + s) * Kk + j) * (ll)(2 * D);
  out[ro + c]     = fn - fc;
  out[ro + D + c] = fc;
}

__global__ void k_maxk(const float* __restrict__ mid, int C, int Bb, int S, int Kk,
                       float* __restrict__ out)
{
  int i = blockIdx.x * blockDim.x + threadIdx.x;
  int total = Bb * C * S;
  if (i >= total) return;
  int s = i % S; int r = i / S; int o = r % C; int b = r / C;
  ll NN = (ll)Bb * S * Kk;
  const float* p = mid + (ll)o * NN + ((ll)b * S + s) * Kk;
  float m = p[0];
  for (int j = 1; j < Kk; ++j) m = fmaxf(m, p[j]);
  out[((ll)b * C + o) * S + s] = m;
}

__global__ void k_softmax_row(float* __restrict__ att, int L)
{
  __shared__ float sm[128];
  ll row = blockIdx.x;
  int t = threadIdx.x;
  float* p = att + row * L;
  float v = (t < L) ? p[t] : -1e30f;
  sm[t] = v; __syncthreads();
  for (int s = 64; s > 0; s >>= 1) { if (t < s) sm[t] = fmaxf(sm[t], sm[t + s]); __syncthreads(); }
  float mx = sm[0]; __syncthreads();
  float e = (t < L) ? __expf(v - mx) : 0.f;
  sm[t] = e; __syncthreads();
  for (int s = 64; s > 0; s >>= 1) { if (t < s) sm[t] += sm[t + s]; __syncthreads(); }
  float sum = sm[0];
  if (t < L) p[t] = e / sum;
}

__global__ void k_att_colnorm(float* __restrict__ att, int L, int total)
{
  int i = blockIdx.x * blockDim.x + threadIdx.x;
  if (i >= total) return;
  int b = i / L, m = i % L;
  float* base = att + (ll)b * L * L + m;
  float s = 0.f;
  for (int l = 0; l < L; ++l) s += base[(ll)l * L];
  float inv = 1.f / (1e-9f + s);
  for (int l = 0; l < L; ++l) base[(ll)l * L] *= inv;
}

__global__ void k_sub_inplace(const float* __restrict__ x, ll xB,
                              float* __restrict__ t, int inner, int Bb)
{
  int i = blockIdx.x * blockDim.x + threadIdx.x;
  if (i >= Bb * inner) return;
  int b = i / inner; int r = i % inner;
  t[i] = x[(ll)b * xB + r] - t[i];
}

__global__ void k_chanmax(float* __restrict__ ybig, int C, int L, int Bb)
{
  int i = blockIdx.x * blockDim.x + threadIdx.x;
  if (i >= Bb * C) return;
  int b = i / C, c = i % C;
  float* src = ybig + ((ll)b * 2 * C + c) * L;
  float m = src[0];
  for (int l = 1; l < L; ++l) m = fmaxf(m, src[l]);
  float* dst = ybig + ((ll)b * 2 * C + C + c) * L;
  for (int l = 0; l < L; ++l) dst[l] = m;
}

__global__ void k_copy_ch(const float* __restrict__ src, ll sB,
                          float* __restrict__ dst, ll dB, ll dOff,
                          int C, int Nn, int Bb)
{
  ll i = (ll)blockIdx.x * blockDim.x + threadIdx.x;
  ll total = (ll)Bb * C * Nn;
  if (i >= total) return;
  int n = (int)(i % Nn);
  ll r = i / Nn;
  int c = (int)(r % C);
  int b = (int)(r / C);
  dst[(ll)b * dB + dOff + (ll)c * Nn + n] = src[(ll)b * sB + (ll)c * Nn + n];
}

__global__ void k_interp3(const float* __restrict__ q,   // [B,Nq,3]
                          const float* __restrict__ src, // [B,S,3]
                          const float* __restrict__ feat,// [B,C2,S]
                          int Nq, int S, int C2,
                          float* __restrict__ dst, ll dB, ll dOff, int Bb)
{
  int i = blockIdx.x * blockDim.x + threadIdx.x;
  if (i >= Bb * Nq) return;
  int b = i / Nq, n = i % Nq;
  const float* Q = q + ((ll)b * Nq + n) * 3;
  float qx = Q[0], qy = Q[1], qz = Q[2];
  float d0 = 1e30f, d1 = 1e30f, d2 = 1e30f;
  int i0 = 0, i1 = 0, i2 = 0;
  const float* P = src + (ll)b * S * 3;
  for (int s = 0; s < S; ++s) {
    float dx = P[s*3]-qx, dy = P[s*3+1]-qy, dz = P[s*3+2]-qz;
    float d = dx*dx + dy*dy + dz*dz;
    if (d < d0)      { d2=d1; i2=i1; d1=d0; i1=i0; d0=d; i0=s; }
    else if (d < d1) { d2=d1; i2=i1; d1=d;  i1=s; }
    else if (d < d2) { d2=d;  i2=s; }
  }
  float w0 = 1.f/(d0+1e-8f), w1 = 1.f/(d1+1e-8f), w2 = 1.f/(d2+1e-8f);
  float ws = w0 + w1 + w2;
  w0 /= ws; w1 /= ws; w2 /= ws;
  const float* F = feat + (ll)b * C2 * S;
  float* D_ = dst + (ll)b * dB + dOff + n;
  for (int c = 0; c < C2; ++c) {
    const float* Fc = F + (ll)c * S;
    D_[(ll)c * Nq] = w0 * Fc[i0] + w1 * Fc[i1] + w2 * Fc[i2];
  }
}

// --------------------------- input index mapping ---------------------------
enum {
  IN_X = 0,
  IN_C1W = 1, IN_G1, IN_B1, IN_C2W, IN_G2, IN_B2,
  IN_LO1_W1 = 7, IN_LO1_G1, IN_LO1_B1, IN_LO1_W2, IN_LO1_G2, IN_LO1_B2,
  IN_LO2_W1 = 13, IN_LO2_G1, IN_LO2_B1, IN_LO2_W2, IN_LO2_G2, IN_LO2_B2,
  IN_PT_C1W = 19, IN_PT_G1, IN_PT_B1, IN_PT_C2W, IN_PT_G2, IN_PT_B2,
  IN_SA_BASE = 25,          // 4 x {wqk, wv, bv, wt, bt, g, b} -> 25..52
  IN_F1W = 53, IN_F1B, IN_FG1, IN_FB1,
  IN_F2W = 57, IN_F2B, IN_FG2, IN_FB2,
  IN_FP2_0_W = 61, IN_FP2_0_CB, IN_FP2_0_G, IN_FP2_0_B,
  IN_FP2_1_W = 65, IN_FP2_1_CB, IN_FP2_1_G, IN_FP2_1_B,
  IN_FP1_0_W = 69, IN_FP1_0_CB, IN_FP1_0_G, IN_FP1_0_B,
  IN_FP1_1_W = 73, IN_FP1_1_CB, IN_FP1_1_G, IN_FP1_1_B,
  IN_C3W = 77, IN_C3B, IN_G3, IN_B3,
  IN_LW = 81, IN_LB = 82
};

// ------------------------------ orchestration ------------------------------
extern "C" void kernel_launch(void* const* d_in, const int* in_sizes, int n_in,
                              void* d_out, int out_size, void* d_ws, size_t ws_size,
                              hipStream_t stream)
{
  (void)in_sizes; (void)n_in; (void)out_size; (void)ws_size;
  const int Bc = 4, Np = 8192;
  auto P = [&](int i) -> const float* { return (const float*)d_in[i]; };
  char* ws = (char*)d_ws;

  const size_t OFF_F0    = 0;          // [B,64,8192]
  const size_t OFF_XYZ1  = 8388608;    // [B,1024,3]
  const size_t OFF_XYZ2  = 8437760;    // [B,128,3]
  const size_t OFF_FIDX1 = 8443904;    // int [B,1024]
  const size_t OFF_FIDX2 = 8460288;    // int [B,128]
  const size_t OFF_KNN1  = 8462336;    // int [B,1024,32]
  const size_t OFF_KNN2  = 8986624;    // int [B,128,32]
  const size_t OFF_F1    = 9052160;    // [B,128,1024]
  const size_t OFF_F2    = 11149312;   // [B,256,128]
  const size_t OFF_X0    = 11673600;   // [B,256,128]
  const size_t OFF_QK    = 12197888;   // [B,64,128]
  const size_t OFF_ATT   = 12328960;   // [B,128,128]
  const size_t OFF_XV    = 12591104;   // [B,256,128]
  const size_t OFF_XR    = 13115392;   // [B,256,128]
  const size_t OFF_YBIG  = 13639680;   // [B,2048,128]
  const size_t OFF_YF1   = 17833984;   // [B,2048,128]
  const size_t OFF_YF2   = 22028288;   // [B,1024,128]
  const size_t SCR       = 24125440;   // reused scratch region base

  float* f0    = (float*)(ws + OFF_F0);
  float* xyz1  = (float*)(ws + OFF_XYZ1);
  float* xyz2  = (float*)(ws + OFF_XYZ2);
  int*   fidx1 = (int*)  (ws + OFF_FIDX1);
  int*   fidx2 = (int*)  (ws + OFF_FIDX2);
  int*   knn1  = (int*)  (ws + OFF_KNN1);
  int*   knn2  = (int*)  (ws + OFF_KNN2);
  float* f1b_  = (float*)(ws + OFF_F1);
  float* f2b_  = (float*)(ws + OFF_F2);
  float* x0    = (float*)(ws + OFF_X0);
  float* qk    = (float*)(ws + OFF_QK);
  float* att   = (float*)(ws + OFF_ATT);
  float* xv    = (float*)(ws + OFF_XV);
  float* xr    = (float*)(ws + OFF_XR);
  float* ybig  = (float*)(ws + OFF_YBIG);
  float* yf1   = (float*)(ws + OFF_YF1);
  float* yf2   = (float*)(ws + OFF_YF2);

  const float* xin = P(IN_X);
  hipStream_t st = stream;

  // ---- stem ----
  float* h = (float*)(ws + SCR);
  gemm(st, P(IN_C1W),0,3,1,  xin,(ll)Np*3,1,3,  h,(ll)64*Np,(ll)Np,1,
       64, Np, 3, Bc, P(IN_G1), P(IN_B1), nullptr, nullptr, 0, 1);
  gemm(st, P(IN_C2W),0,64,1, h,(ll)64*Np,(ll)Np,1, f0,(ll)64*Np,(ll)Np,1,
       64, Np, 64, Bc, P(IN_G2), P(IN_B2), nullptr, nullptr, 0, 1);

  // ---- level 1 ----
  k_fps<<<Bc, 256, 0, st>>>(xin, Np, 1024, fidx1);
  k_gather_xyz<<<(Bc*1024+255)/256, 256, 0, st>>>(xin, fidx1, Np, 1024, Bc, xyz1);
  k_knn<<<dim3(1024, Bc), 256, 0, st>>>(xin, xyz1, Np, 1024, 32, knn1);

  float* grp1  = (float*)(ws + SCR);            // [131072,128]
  float* mid1a = (float*)(ws + SCR + 67108864); // [128,131072]
  float* mid1b = (float*)(ws + SCR);            // overwrites grp1
  {
    ll total = (ll)Bc*1024*32*64;
    k_group<<<(unsigned)((total+255)/256), 256, 0, st>>>(f0, fidx1, knn1, Np, 1024, 32, 64, total, grp1);
  }
  gemm(st, P(IN_LO1_W1),0,128,1, grp1,0,1,128, mid1a,0,131072,1,
       128, 131072, 128, 1, P(IN_LO1_G1), P(IN_LO1_B1), nullptr, nullptr, 0, 1);
  gemm(st, P(IN_LO1_W2),0,128,1, mid1a,0,131072,1, mid1b,0,131072,1,
       128, 131072, 128, 1, P(IN_LO1_G2), P(IN_LO1_B2), nullptr, nullptr, 0, 1);
  k_maxk<<<(Bc*128*1024+255)/256, 256, 0, st>>>(mid1b, 128, Bc, 1024, 32, f1b_);

  // ---- level 2 ----
  k_fps<<<Bc, 256, 0, st>>>(xyz1, 1024, 128, fidx2);
  k_gather_xyz<<<(Bc*128+255)/256, 256, 0, st>>>(xyz1, fidx2, 1024, 128, Bc, xyz2);
  k_knn<<<dim3(128, Bc), 256, 0, st>>>(xyz1, xyz2, 1024, 128, 32, knn2);

  float* grp2  = (float*)(ws + SCR);             // [16384,256]
  float* mid2a = (float*)(ws + SCR + 16777216);  // [256,16384]
  float* mid2b = (float*)(ws + SCR + 33554432);  // [256,16384]
  {
    ll total = (ll)Bc*128*32*128;
    k_group<<<(unsigned)((total+255)/256), 256, 0, st>>>(f1b_, fidx2, knn2, 1024, 128, 32, 128, total, grp2);
  }
  gemm(st, P(IN_LO2_W1),0,256,1, grp2,0,1,256, mid2a,0,16384,1,
       256, 16384, 256, 1, P(IN_LO2_G1), P(IN_LO2_B1), nullptr, nullptr, 0, 1);
  gemm(st, P(IN_LO2_W2),0,256,1, mid2a,0,16384,1, mid2b,0,16384,1,
       256, 16384, 256, 1, P(IN_LO2_G2), P(IN_LO2_B2), nullptr, nullptr, 0, 1);
  k_maxk<<<(Bc*256*128+255)/256, 256, 0, st>>>(mid2b, 256, Bc, 128, 32, f2b_);

  // ---- stacked attention (L=128, C=256, Cq=64) ----
  const int L = 128;
  gemm(st, P(IN_PT_C1W),0,256,1, f2b_,(ll)256*L,(ll)L,1, xv,(ll)256*L,(ll)L,1,
       256, L, 256, Bc, P(IN_PT_G1), P(IN_PT_B1), nullptr, nullptr, 0, 1);
  gemm(st, P(IN_PT_C2W),0,256,1, xv,(ll)256*L,(ll)L,1, x0,(ll)256*L,(ll)L,1,
       256, L, 256, Bc, P(IN_PT_G2), P(IN_PT_B2), nullptr, nullptr, 0, 1);

  const float* xcur = x0;
  ll xb = (ll)256 * L;
  for (int i = 0; i < 4; ++i) {
    const float* wqk = P(IN_SA_BASE + 7*i + 0);
    const float* wv  = P(IN_SA_BASE + 7*i + 1);
    const float* bv  = P(IN_SA_BASE + 7*i + 2);
    const float* wt  = P(IN_SA_BASE + 7*i + 3);
    const float* bt  = P(IN_SA_BASE + 7*i + 4);
    const float* gg  = P(IN_SA_BASE + 7*i + 5);
    const float* bb  = P(IN_SA_BASE + 7*i + 6);
    gemm(st, wqk,0,256,1, xcur,xb,(ll)L,1, qk,(ll)64*L,(ll)L,1,
         64, L, 256, Bc, nullptr, nullptr, nullptr, nullptr, 0, 0);
    gemm(st, qk,(ll)64*L,1,(ll)L, qk,(ll)64*L,(ll)L,1, att,(ll)L*L,(ll)L,1,
         L, L, 64, Bc, nullptr, nullptr, nullptr, nullptr, 0, 0);
    k_softmax_row<<<Bc*L, 128, 0, st>>>(att, L);
    k_att_colnorm<<<(Bc*L+255)/256, 256, 0, st>>>(att, L, Bc*L);
    gemm(st, wv,0,256,1, xcur,xb,(ll)L,1, xv,(ll)256*L,(ll)L,1,
         256, L, 256, Bc, nullptr, nullptr, bv, nullptr, 0, 0);
    gemm(st, xv,(ll)256*L,(ll)L,1, att,(ll)L*L,(ll)L,1, xr,(ll)256*L,(ll)L,1,
         256, L, L, Bc, nullptr, nullptr, nullptr, nullptr, 0, 0);
    k_sub_inplace<<<(Bc*256*L+255)/256, 256, 0, st>>>(xcur, xb, xr, 256*L, Bc);
    float* yslab = ybig + (ll)i * 256 * L;
    gemm(st, wt,0,256,1, xr,(ll)256*L,(ll)L,1, yslab,(ll)2048*L,(ll)L,1,
         256, L, 256, Bc, gg, bb, bt, xcur, xb, 1);
    xcur = yslab;
    xb = (ll)2048 * L;
  }

  // ---- global max concat + f1/f2 ----
  k_chanmax<<<(Bc*1024+255)/256, 256, 0, st>>>(ybig, 1024, L, Bc);
  gemm(st, P(IN_F1W),0,2048,1, ybig,(ll)2048*L,(ll)L,1, yf1,(ll)2048*L,(ll)L,1,
       2048, L, 2048, Bc, P(IN_FG1), P(IN_FB1), P(IN_F1B), nullptr, 0, 1);
  gemm(st, P(IN_F2W),0,2048,1, yf1,(ll)2048*L,(ll)L,1, yf2,(ll)1024*L,(ll)L,1,
       1024, L, 2048, Bc, P(IN_FG2), P(IN_FB2), P(IN_F2B), nullptr, 0, 1);

  // ---- feat_prop 2 ----
  float* fp2cat = (float*)(ws + SCR);             // [B,1152,1024]
  float* fp2mid = (float*)(ws + SCR + 18874368);  // [B,1024,1024]
  float* fp2out = (float*)(ws + SCR + 35651584);  // [B,512,1024]
  k_copy_ch<<<(unsigned)(((ll)Bc*128*1024+255)/256), 256, 0, st>>>(
      f1b_, (ll)128*1024, fp2cat, (ll)1152*1024, 0, 128, 1024, Bc);
  k_interp3<<<(Bc*1024+127)/128, 128, 0, st>>>(
      xyz1, xyz2, yf2, 1024, 128, 1024, fp2cat, (ll)1152*1024, (ll)128*1024, Bc);
  gemm(st, P(IN_FP2_0_W),0,1152,1, fp2cat,(ll)1152*1024,1024,1, fp2mid,(ll)1024*1024,1024,1,
       1024, 1024, 1152, Bc, P(IN_FP2_0_G), P(IN_FP2_0_B), P(IN_FP2_0_CB), nullptr, 0, 1);
  gemm(st, P(IN_FP2_1_W),0,1024,1, fp2mid,(ll)1024*1024,1024,1, fp2out,(ll)512*1024,1024,1,
       512, 1024, 1024, Bc, P(IN_FP2_1_G), P(IN_FP2_1_B), P(IN_FP2_1_CB), nullptr, 0, 1);

  // ---- feat_prop 1 ----
  float* fp1cat = (float*)(ws + SCR + 50331648);   // [B,576,8192]
  float* fp1mid = (float*)(ws + SCR + 134217728);  // [B,512,8192]
  float* fp1out = (float*)(ws + SCR);              // [B,256,8192]
  k_copy_ch<<<(unsigned)(((ll)Bc*64*Np+255)/256), 256, 0, st>>>(
      f0, (ll)64*Np, fp1cat, (ll)576*Np, 0, 64, Np, Bc);
  k_interp3<<<(Bc*Np+127)/128, 128, 0, st>>>(
      xin, xyz1, fp2out, Np, 1024, 512, fp1cat, (ll)576*Np, (ll)64*Np, Bc);
  gemm(st, P(IN_FP1_0_W),0,576,1, fp1cat,(ll)576*Np,(ll)Np,1, fp1mid,(ll)512*Np,(ll)Np,1,
       512, Np, 576, Bc, P(IN_FP1_0_G), P(IN_FP1_0_B), P(IN_FP1_0_CB), nullptr, 0, 1);
  gemm(st, P(IN_FP1_1_W),0,512,1, fp1mid,(ll)512*Np,(ll)Np,1, fp1out,(ll)256*Np,(ll)Np,1,
       256, Np, 512, Bc, P(IN_FP1_1_G), P(IN_FP1_1_B), P(IN_FP1_1_CB), nullptr, 0, 1);

  // ---- head ----
  float* c3out = (float*)(ws + SCR + 35651584);    // [B,128,8192]
  gemm(st, P(IN_C3W),0,256,1, fp1out,(ll)256*Np,(ll)Np,1, c3out,(ll)128*Np,(ll)Np,1,
       128, Np, 256, Bc, P(IN_G3), P(IN_B3), P(IN_C3B), nullptr, 0, 1);
  gemm(st, P(IN_LW),0,128,1, c3out,(ll)128*Np,(ll)Np,1,
       (float*)d_out,(ll)Np*8,1,8,
       8, Np, 128, Bc, nullptr, nullptr, P(IN_LB), nullptr, 0, 0);
}
